// TTTPolicy_66769561583656
// MI455X (gfx1250) — matrix operand order repaired and checked by reading, not compiled, not run on hardware
//
#include <hip/hip_runtime.h>

// ---------------- model dims ----------------
#define K_B    64
#define K_H    512
#define K_NH   8
#define K_HD   64
#define K_NL   8
#define K_IN   1376
#define K_MBS  16
#define K_SEQ  32
#define K_TOK  (K_B * K_SEQ)     // 2048 tokens
#define K_OBS  264
#define K_EPS  1e-6f

// ---------------- WMMA types (gfx1250, wave32) ----------------
typedef __attribute__((ext_vector_type(16))) __bf16          bf16x16;
typedef __attribute__((ext_vector_type(8)))  float           f32x8;
typedef __attribute__((ext_vector_type(8)))  unsigned short  u16x8;

union Frag { u16x8 h[2]; bf16x16 v; };

__device__ __forceinline__ unsigned short f2bf(float f) {
  unsigned u = __float_as_uint(f);
  u += 0x7FFFu + ((u >> 16) & 1u);           // round-to-nearest-even
  return (unsigned short)(u >> 16);
}
__device__ __forceinline__ float bf2f(unsigned short h) {
  return __uint_as_float(((unsigned)h) << 16);
}

// =====================================================================
// Weight convert + transpose: W (K x N, fp32) -> Wt (N x K, bf16)
// so the WMMA B operand load is contiguous per lane.
// =====================================================================
__global__ void w_to_bf16_t(const float* __restrict__ W,
                            unsigned short* __restrict__ Wt, int K, int N) {
  int total = K * N;
  for (int idx = blockIdx.x * blockDim.x + threadIdx.x; idx < total;
       idx += gridDim.x * blockDim.x) {
    int n = idx / K;
    int k = idx - n * K;
    Wt[idx] = f2bf(W[(size_t)k * N + n]);
  }
}

// =====================================================================
// GEMM: C(MxN, f32) = A(MxK, bf16 row-major) * Bt(NxK, bf16)^T  [+ Res]
// Block = 2 waves (64 thr), block tile 128x32; wave tile 64x32:
//   4 A-frags x 2 B-frags -> 8 WMMAs per K-step (1.33 WMMA / frag-load).
// A fragment layout (16-bit A 16x32):   lane l: m = l&15, hi = l>>4,
//   element j -> K = k0 + hi*8 + j + (j>=8 ? 8 : 0)   (two 16B loads)
// B fragment layout (16-bit B 32x16):   lane l: n = l&15,
//   element j -> K = k0 + hi*8 + j + (j>=8 ? 8 : 0)   (two 16B loads)
// C/D f32 16x16: lane l: n = l&15, vgpr r -> m = hi*8 + r.
// =====================================================================
__global__ __launch_bounds__(64) void gemm_bf16(
    const unsigned short* __restrict__ A,
    const unsigned short* __restrict__ Bt,
    const float* Res,        // optional residual (may alias C); null = none
    float* C,
    int M, int N, int K) {
  const int wave = threadIdx.x >> 5;
  const int lane = threadIdx.x & 31;
  const int l16  = lane & 15;
  const int hi   = lane >> 4;
  const int m0   = blockIdx.x * 128 + wave * 64;
  const int n0   = blockIdx.y * 32;

  f32x8 acc[4][2];
#pragma unroll
  for (int i = 0; i < 4; ++i)
#pragma unroll
    for (int j = 0; j < 2; ++j) acc[i][j] = (f32x8){};

  const unsigned short* aBase = A + (size_t)(m0 + l16) * K + hi * 8;
  const unsigned short* bBase = Bt + (size_t)(n0 + l16) * K + hi * 8;
  const size_t aRowStride = (size_t)16 * K;   // 16 rows per fragment

  for (int k0 = 0; k0 < K; k0 += 32) {
    Frag a[4], b[2];
#pragma unroll
    for (int t = 0; t < 4; ++t) {
      const unsigned short* ap = aBase + (size_t)t * aRowStride + k0;
      a[t].h[0] = *(const u16x8*)(ap);
      a[t].h[1] = *(const u16x8*)(ap + 16);
    }
#pragma unroll
    for (int t = 0; t < 2; ++t) {
      const unsigned short* bp = bBase + (size_t)t * aRowStride + k0;
      b[t].h[0] = *(const u16x8*)(bp);
      b[t].h[1] = *(const u16x8*)(bp + 16);
    }
#pragma unroll
    for (int i = 0; i < 4; ++i)
#pragma unroll
      for (int j = 0; j < 2; ++j)
        acc[i][j] = __builtin_amdgcn_wmma_f32_16x16x32_bf16(
            false, a[i].v, false, b[j].v, (short)0, acc[i][j], false, false);
  }

  const int nc = n0 + l16;
#pragma unroll
  for (int i = 0; i < 4; ++i) {
#pragma unroll
    for (int j = 0; j < 2; ++j) {
#pragma unroll
      for (int r = 0; r < 8; ++r) {
        int m = m0 + i * 16 + hi * 8 + r;
        size_t idx = (size_t)m * N + nc + j * 16;
        float v = acc[i][j][r];
        if (Res) v += Res[idx];
        C[idx] = v;
      }
    }
  }
}

// =====================================================================
// Embedding: x[b,0,:] = state@emb_s_w + b ; x[b,1+t,:] = traj@emb_t_w + b
// =====================================================================
__global__ void embed_kernel(const float* __restrict__ obs,
                             const float* __restrict__ esw, const float* __restrict__ esb,
                             const float* __restrict__ etw, const float* __restrict__ etb,
                             float* __restrict__ x) {
  int idx = blockIdx.x * blockDim.x + threadIdx.x;
  if (idx >= K_TOK * K_H) return;
  int h  = idx & (K_H - 1);
  int tk = idx >> 9;               // token index
  int b  = tk / K_SEQ;
  int t  = tk - b * K_SEQ;
  float acc;
  if (t == 0) {
    acc = esb[h];
    const float* o = obs + (size_t)b * K_OBS;
    for (int s = 0; s < 16; ++s) acc += o[s] * esw[s * K_H + h];
  } else {
    acc = etb[h];
    const float* o = obs + (size_t)b * K_OBS + 16 + (t - 1) * 8;
    for (int r = 0; r < 8; ++r) acc += o[r] * etw[r * K_H + h];
  }
  x[idx] = acc;
}

// =====================================================================
// RMSNorm row (H=512) -> bf16
// =====================================================================
__global__ __launch_bounds__(256) void rms_bf16_kernel(
    const float* __restrict__ x, const float* __restrict__ w,
    unsigned short* __restrict__ out) {
  int row = blockIdx.x;
  const float* xr = x + (size_t)row * K_H;
  float ss = 0.f;
  for (int i = threadIdx.x; i < K_H; i += 256) { float v = xr[i]; ss += v * v; }
  __shared__ float red[256];
  red[threadIdx.x] = ss; __syncthreads();
  for (int s = 128; s > 0; s >>= 1) {
    if (threadIdx.x < s) red[threadIdx.x] += red[threadIdx.x + s];
    __syncthreads();
  }
  float rs = rsqrtf(red[0] * (1.f / K_H) + K_EPS);
  for (int i = threadIdx.x; i < K_H; i += 256)
    out[(size_t)row * K_H + i] = f2bf(xr[i] * rs * w[i]);
}

// =====================================================================
// LayerNorm row (H=512) -> bf16 (post-TTT norm)
// =====================================================================
__global__ __launch_bounds__(256) void ln_bf16_kernel(
    const float* __restrict__ x, const float* __restrict__ w,
    const float* __restrict__ b, unsigned short* __restrict__ out) {
  int row = blockIdx.x;
  const float* xr = x + (size_t)row * K_H;
  float s = 0.f, q = 0.f;
  for (int i = threadIdx.x; i < K_H; i += 256) { float v = xr[i]; s += v; q += v * v; }
  __shared__ float rs_[256], rq_[256];
  rs_[threadIdx.x] = s; rq_[threadIdx.x] = q; __syncthreads();
  for (int st = 128; st > 0; st >>= 1) {
    if (threadIdx.x < st) { rs_[threadIdx.x] += rs_[threadIdx.x + st];
                            rq_[threadIdx.x] += rq_[threadIdx.x + st]; }
    __syncthreads();
  }
  float mu  = rs_[0] * (1.f / K_H);
  float var = rq_[0] * (1.f / K_H) - mu * mu;
  float rstd = rsqrtf(var + K_EPS);
  for (int i = threadIdx.x; i < K_H; i += 256)
    out[(size_t)row * K_H + i] = f2bf((xr[i] - mu) * rstd * w[i] + b[i]);
}

// =====================================================================
// lr = sigmoid(xn . lr_w[n] + lr_b[n])  -> (B, NH, L)
// =====================================================================
__global__ void lr_kernel(const unsigned short* __restrict__ xn,
                          const float* __restrict__ lr_w,
                          const float* __restrict__ lr_b,
                          float* __restrict__ lrout) {
  int idx = blockIdx.x * blockDim.x + threadIdx.x;      // B*NH*L = 16384
  if (idx >= K_B * K_NH * K_SEQ) return;
  int l = idx & 31;
  int n = (idx >> 5) & 7;
  int b = idx >> 8;
  const unsigned short* xr = xn + ((size_t)b * K_SEQ + l) * K_H;
  const float* wr = lr_w + n * K_H;
  float acc = lr_b[n];
  for (int h = 0; h < K_H; ++h) acc += bf2f(xr[h]) * wr[h];
  lrout[idx] = 1.f / (1.f + expf(-acc));
}

// =====================================================================
// TTT inner scan: one block per (batch, head), 256 threads.
// W1 (64x64) fast-weight state lives in LDS; RoPE fused at Q/K load.
// L=32 -> 2 sequential mini-batches of 16.
// =====================================================================
__global__ __launch_bounds__(256) void ttt_kernel(
    const float* __restrict__ qg, const float* __restrict__ kg,
    const float* __restrict__ vg, const float* __restrict__ lrbuf,
    const float* __restrict__ W1p, const float* __restrict__ b1p,
    const float* __restrict__ gp,  const float* __restrict__ bp,
    const float* __restrict__ tokp, float* __restrict__ outg) {
  const int bh = blockIdx.x;
  const int b  = bh >> 3;
  const int n  = bh & 7;
  const int tid = threadIdx.x;
  const int i  = tid >> 4;       // mini-batch row 0..15
  const int tx = tid & 15;       // covers f = tx*4 .. tx*4+3

  __shared__ float W1[K_HD][K_HD];
  __shared__ float b1s[K_HD], gv[K_HD], bv[K_HD];
  __shared__ float Qs[K_MBS][K_HD], Ks[K_MBS][K_HD], Vs[K_MBS][K_HD];
  __shared__ float Gr[K_MBS][K_HD];
  __shared__ float At[K_MBS][K_MBS];
  __shared__ float redA[16][16], redB[16][16];
  __shared__ float mu_[16], rstd_[16], s1_[16], s2_[16];
  __shared__ float ev[K_MBS], tokv[K_MBS];

  for (int idx = tid; idx < K_HD * K_HD; idx += 256)
    W1[idx >> 6][idx & 63] = W1p[n * K_HD * K_HD + idx];
  if (tid < K_HD) {
    b1s[tid] = b1p[n * K_HD + tid];
    gv[tid]  = gp[n * K_HD + tid];
    bv[tid]  = bp[n * K_HD + tid];
  }
  if (tid < K_MBS) tokv[tid] = fmaxf(1.0f / (float)(tid + 1) + tokp[tid], 0.0f);
  __syncthreads();

  for (int mb = 0; mb < 2; ++mb) {
    const int l = mb * 16 + i;
    // ---- load Q, K (with RoPE), V, lr ----
    {
      size_t base = ((size_t)b * K_SEQ + l) * K_H + n * K_HD;
      for (int c = 0; c < 4; ++c) {
        int f  = tx * 4 + c;
        int fr = f & 31;
        float inv = expf(-(float)fr * (1.0f / 32.0f) * 9.210340371976184f); // 10000^(-fr/32)
        float th = (float)l * inv;
        float cs = cosf(th), sn = sinf(th);
        int fo = (f < 32) ? f + 32 : f - 32;
        float q1 = qg[base + f], q2 = qg[base + fo];
        float k1 = kg[base + f], k2 = kg[base + fo];
        Qs[i][f] = (f < 32) ? q1 * cs - q2 * sn : q1 * cs + q2 * sn;
        Ks[i][f] = (f < 32) ? k1 * cs - k2 * sn : k1 * cs + k2 * sn;
        Vs[i][f] = vg[base + f];
      }
      if (tid < K_MBS)
        ev[tid] = (1.0f / (float)K_HD) *
                  lrbuf[((size_t)b * K_NH + n) * K_SEQ + mb * 16 + tid];  // BASE_LR=1
    }
    __syncthreads();

    // ---- Z1 = K @ W1 + b1, with LN stats ----
    float zloc[4];
    float ssum = 0.f, ssq = 0.f;
    for (int c = 0; c < 4; ++c) {
      int f = tx * 4 + c;
      float acc = b1s[f];
      for (int cc = 0; cc < K_HD; ++cc) acc += Ks[i][cc] * W1[cc][f];
      zloc[c] = acc; ssum += acc; ssq += acc * acc;
    }
    redA[i][tx] = ssum; redB[i][tx] = ssq;
    __syncthreads();
    if (tid < 16) {
      float s = 0.f, q = 0.f;
      for (int t = 0; t < 16; ++t) { s += redA[tid][t]; q += redB[tid][t]; }
      float m = s * (1.f / K_HD);
      mu_[tid] = m; rstd_[tid] = rsqrtf(q * (1.f / K_HD) - m * m + K_EPS);
    }
    __syncthreads();

    // ---- grad = ln_l2_bwd(Z1, V-K, g, b) ----
    float gxh[4], xh[4];
    ssum = 0.f; ssq = 0.f;
    for (int c = 0; c < 4; ++c) {
      int f = tx * 4 + c;
      xh[c] = (zloc[c] - mu_[i]) * rstd_[i];
      float tgt = Vs[i][f] - Ks[i][f];
      gxh[c] = (gv[f] * xh[c] + bv[f] - tgt) * gv[f];
      ssum += gxh[c]; ssq += gxh[c] * xh[c];
    }
    redA[i][tx] = ssum; redB[i][tx] = ssq;
    __syncthreads();
    if (tid < 16) {
      float s = 0.f, q = 0.f;
      for (int t = 0; t < 16; ++t) { s += redA[tid][t]; q += redB[tid][t]; }
      s1_[tid] = s; s2_[tid] = q;
    }
    __syncthreads();
    for (int c = 0; c < 4; ++c) {
      int f = tx * 4 + c;
      Gr[i][f] = ((float)K_HD * gxh[c] - s1_[i] - xh[c] * s2_[i]) *
                 rstd_[i] * (1.f / K_HD);
    }
    // ---- attn = (Q @ K^T) * tril ----
    {
      float a = 0.f;
      if (tx <= i) {
        for (int cc = 0; cc < K_HD; ++cc) a += Qs[i][cc] * Ks[tx][cc];
      }
      At[i][tx] = a;
    }
    __syncthreads();

    // ---- Z1_bar = Q@W1 + b1 - sum_j e_ij (attn_ij + 1) grad_j ----
    ssum = 0.f; ssq = 0.f;
    for (int c = 0; c < 4; ++c) {
      int f = tx * 4 + c;
      float acc = b1s[f];
      for (int cc = 0; cc < K_HD; ++cc) acc += Qs[i][cc] * W1[cc][f];
      float ti = tokv[i];
      for (int j = 0; j <= i; ++j)
        acc -= ti * ev[j] * (At[i][j] + 1.f) * Gr[j][f];
      zloc[c] = acc; ssum += acc; ssq += acc * acc;
    }
    redA[i][tx] = ssum; redB[i][tx] = ssq;
    __syncthreads();
    if (tid < 16) {
      float s = 0.f, q = 0.f;
      for (int t = 0; t < 16; ++t) { s += redA[tid][t]; q += redB[tid][t]; }
      float m = s * (1.f / K_HD);
      mu_[tid] = m; rstd_[tid] = rsqrtf(q * (1.f / K_HD) - m * m + K_EPS);
    }
    __syncthreads();
    // ---- out = Q + ln(Z1_bar) ----
    for (int c = 0; c < 4; ++c) {
      int f = tx * 4 + c;
      float o = Qs[i][f] + (zloc[c] - mu_[i]) * rstd_[i] * gv[f] + bv[f];
      outg[((size_t)b * K_SEQ + l) * K_H + n * K_HD + f] = o;
    }
    __syncthreads();
    // ---- state update: W1 -= K^T diag(last) grad ; b1 -= last . grad ----
    const float tlast = tokv[K_MBS - 1];
    for (int idx = tid; idx < K_HD * K_HD; idx += 256) {
      int cc = idx >> 6, f = idx & 63;
      float acc = 0.f;
      for (int j = 0; j < K_MBS; ++j) acc += ev[j] * Ks[j][cc] * Gr[j][f];
      W1[cc][f] -= tlast * acc;
    }
    if (tid < K_HD) {
      float acc = 0.f;
      for (int j = 0; j < K_MBS; ++j) acc += ev[j] * Gr[j][tid];
      b1s[tid] -= tlast * acc;
    }
    __syncthreads();
  }
}

// =====================================================================
// SwiGLU elementwise: inter = bf16( silu(g) * u )
// =====================================================================
__global__ void silu_mul_kernel(const float* __restrict__ g,
                                const float* __restrict__ u,
                                unsigned short* __restrict__ out, int total) {
  for (int idx = blockIdx.x * blockDim.x + threadIdx.x; idx < total;
       idx += gridDim.x * blockDim.x) {
    float gv = g[idx];
    float s  = gv / (1.f + expf(-gv));
    out[idx] = f2bf(s * u[idx]);
  }
}

// =====================================================================
// Head: rms(x[b, SEQ-1, :], final_nw) @ act_w + act_b -> tanh squash
// =====================================================================
__global__ __launch_bounds__(256) void head_kernel(
    const float* __restrict__ x, const float* __restrict__ fnw,
    const float* __restrict__ aw, const float* __restrict__ ab,
    const float* __restrict__ ahi, const float* __restrict__ alo,
    float* __restrict__ out) {
  int b = blockIdx.x;
  const float* xr = x + ((size_t)b * K_SEQ + (K_SEQ - 1)) * K_H;
  __shared__ float nrm[K_H];
  __shared__ float red[256], r0[256], r1[256];
  float ss = 0.f;
  for (int i = threadIdx.x; i < K_H; i += 256) { float v = xr[i]; ss += v * v; }
  red[threadIdx.x] = ss; __syncthreads();
  for (int s = 128; s > 0; s >>= 1) {
    if (threadIdx.x < s) red[threadIdx.x] += red[threadIdx.x + s];
    __syncthreads();
  }
  float rs = rsqrtf(red[0] * (1.f / K_H) + K_EPS);
  for (int i = threadIdx.x; i < K_H; i += 256) nrm[i] = xr[i] * rs * fnw[i];
  __syncthreads();
  float p0 = 0.f, p1 = 0.f;
  for (int i = threadIdx.x; i < K_H; i += 256) {
    p0 += nrm[i] * aw[i * 2 + 0];
    p1 += nrm[i] * aw[i * 2 + 1];
  }
  r0[threadIdx.x] = p0; r1[threadIdx.x] = p1; __syncthreads();
  for (int s = 128; s > 0; s >>= 1) {
    if (threadIdx.x < s) { r0[threadIdx.x] += r0[threadIdx.x + s];
                           r1[threadIdx.x] += r1[threadIdx.x + s]; }
    __syncthreads();
  }
  if (threadIdx.x == 0) {
    float a0 = r0[0] + ab[0], a1 = r1[0] + ab[1];
    out[b * 2 + 0] = 0.5f * (ahi[0] - alo[0]) * tanhf(a0) + 0.5f * (ahi[0] + alo[0]);
    out[b * 2 + 1] = 0.5f * (ahi[1] - alo[1]) * tanhf(a1) + 0.5f * (ahi[1] + alo[1]);
  }
}

// =====================================================================
// Host orchestration
// =====================================================================
extern "C" void kernel_launch(void* const* d_in, const int* in_sizes, int n_in,
                              void* d_out, int out_size, void* d_ws, size_t ws_size,
                              hipStream_t stream) {
  (void)in_sizes; (void)n_in; (void)out_size; (void)ws_size;
  // ---- inputs, setup_inputs() dict insertion order ----
  const float* obs      = (const float*)d_in[0];
  const float* wq       = (const float*)d_in[1];   // (NL,H,H)
  const float* wk       = (const float*)d_in[2];
  const float* wv       = (const float*)d_in[3];
  const float* wo       = (const float*)d_in[4];
  const float* W1       = (const float*)d_in[5];   // (NL,NH,HD,HD)
  const float* b1       = (const float*)d_in[6];   // (NL,NH,1,HD)
  const float* lr_w     = (const float*)d_in[7];   // (NL,NH,H)
  const float* lr_b     = (const float*)d_in[8];   // (NL,NH)
  const float* tok_idx  = (const float*)d_in[9];   // (NL,MB)
  const float* ttt_g    = (const float*)d_in[10];  // (NL,NH,HD)
  const float* ttt_b    = (const float*)d_in[11];
  const float* post_w   = (const float*)d_in[12];  // (NL,H)
  const float* post_b   = (const float*)d_in[13];
  const float* seq_nw   = (const float*)d_in[14];
  const float* ffn_nw   = (const float*)d_in[15];
  const float* gate     = (const float*)d_in[16];  // (NL,H,INTER)
  const float* up       = (const float*)d_in[17];
  const float* down     = (const float*)d_in[18];  // (NL,INTER,H)
  const float* final_nw = (const float*)d_in[19];
  const float* emb_s_w  = (const float*)d_in[20];
  const float* emb_s_b  = (const float*)d_in[21];
  const float* emb_t_w  = (const float*)d_in[22];
  const float* emb_t_b  = (const float*)d_in[23];
  const float* act_w    = (const float*)d_in[24];
  const float* act_b    = (const float*)d_in[25];
  const float* act_high = (const float*)d_in[26];
  const float* act_low  = (const float*)d_in[27];
  float* out = (float*)d_out;

  // ---- workspace carve ----
  char* p = (char*)d_ws;
  auto carve = [&](size_t bytes) -> void* {
    void* r = (void*)p;
    p += (bytes + 255) & ~(size_t)255;
    return r;
  };
  float*          x      = (float*)carve((size_t)K_TOK * K_H * 4);
  unsigned short* xn     = (unsigned short*)carve((size_t)K_TOK * K_H * 2);
  float*          qb     = (float*)carve((size_t)K_TOK * K_H * 4);
  float*          kb     = (float*)carve((size_t)K_TOK * K_H * 4);
  float*          vb     = (float*)carve((size_t)K_TOK * K_H * 4);
  float*          tttout = (float*)carve((size_t)K_TOK * K_H * 4);
  unsigned short* lnout  = (unsigned short*)carve((size_t)K_TOK * K_H * 2);
  float*          lrbuf  = (float*)carve((size_t)K_B * K_NH * K_SEQ * 4);
  float*          gbuf   = (float*)carve((size_t)K_TOK * K_IN * 4);
  float*          ubuf   = (float*)carve((size_t)K_TOK * K_IN * 4);
  unsigned short* interb = (unsigned short*)carve((size_t)K_TOK * K_IN * 2);
  unsigned short* wq_t   = (unsigned short*)carve((size_t)K_H * K_H * 2);
  unsigned short* wk_t   = (unsigned short*)carve((size_t)K_H * K_H * 2);
  unsigned short* wv_t   = (unsigned short*)carve((size_t)K_H * K_H * 2);
  unsigned short* wo_t   = (unsigned short*)carve((size_t)K_H * K_H * 2);
  unsigned short* gate_t = (unsigned short*)carve((size_t)K_H * K_IN * 2);
  unsigned short* up_t   = (unsigned short*)carve((size_t)K_H * K_IN * 2);
  unsigned short* down_t = (unsigned short*)carve((size_t)K_IN * K_H * 2);

  // ---- embedding ----
  embed_kernel<<<dim3((K_TOK * K_H + 255) / 256), dim3(256), 0, stream>>>(
      obs, emb_s_w, emb_s_b, emb_t_w, emb_t_b, x);

  const int nHH = K_H * K_H, nHI = K_H * K_IN;
  for (int L = 0; L < K_NL; ++L) {
    // -- per-layer bf16 transposed weights --
    w_to_bf16_t<<<dim3((nHH + 255) / 256), dim3(256), 0, stream>>>(wq + (size_t)L * nHH, wq_t, K_H, K_H);
    w_to_bf16_t<<<dim3((nHH + 255) / 256), dim3(256), 0, stream>>>(wk + (size_t)L * nHH, wk_t, K_H, K_H);
    w_to_bf16_t<<<dim3((nHH + 255) / 256), dim3(256), 0, stream>>>(wv + (size_t)L * nHH, wv_t, K_H, K_H);
    w_to_bf16_t<<<dim3((nHH + 255) / 256), dim3(256), 0, stream>>>(wo + (size_t)L * nHH, wo_t, K_H, K_H);
    w_to_bf16_t<<<dim3((nHI + 255) / 256), dim3(256), 0, stream>>>(gate + (size_t)L * nHI, gate_t, K_H, K_IN);
    w_to_bf16_t<<<dim3((nHI + 255) / 256), dim3(256), 0, stream>>>(up + (size_t)L * nHI, up_t, K_H, K_IN);
    w_to_bf16_t<<<dim3((nHI + 255) / 256), dim3(256), 0, stream>>>(down + (size_t)L * nHI, down_t, K_IN, K_H);

    // -- TTT block --
    rms_bf16_kernel<<<dim3(K_TOK), dim3(256), 0, stream>>>(x, seq_nw + L * K_H, xn);
    gemm_bf16<<<dim3(K_TOK / 128, K_H / 32), dim3(64), 0, stream>>>(xn, wq_t, nullptr, qb, K_TOK, K_H, K_H);
    gemm_bf16<<<dim3(K_TOK / 128, K_H / 32), dim3(64), 0, stream>>>(xn, wk_t, nullptr, kb, K_TOK, K_H, K_H);
    gemm_bf16<<<dim3(K_TOK / 128, K_H / 32), dim3(64), 0, stream>>>(xn, wv_t, nullptr, vb, K_TOK, K_H, K_H);
    lr_kernel<<<dim3((K_B * K_NH * K_SEQ + 255) / 256), dim3(256), 0, stream>>>(
        xn, lr_w + (size_t)L * K_NH * K_H, lr_b + L * K_NH, lrbuf);
    ttt_kernel<<<dim3(K_B * K_NH), dim3(256), 0, stream>>>(
        qb, kb, vb, lrbuf,
        W1 + (size_t)L * K_NH * K_HD * K_HD, b1 + (size_t)L * K_NH * K_HD,
        ttt_g + (size_t)L * K_NH * K_HD, ttt_b + (size_t)L * K_NH * K_HD,
        tok_idx + L * K_MBS, tttout);
    ln_bf16_kernel<<<dim3(K_TOK), dim3(256), 0, stream>>>(
        tttout, post_w + L * K_H, post_b + L * K_H, lnout);
    gemm_bf16<<<dim3(K_TOK / 128, K_H / 32), dim3(64), 0, stream>>>(
        lnout, wo_t, x, x, K_TOK, K_H, K_H);   // residual add into x

    // -- SwiGLU FFN --
    rms_bf16_kernel<<<dim3(K_TOK), dim3(256), 0, stream>>>(x, ffn_nw + L * K_H, xn);
    gemm_bf16<<<dim3(K_TOK / 128, K_IN / 32), dim3(64), 0, stream>>>(xn, gate_t, nullptr, gbuf, K_TOK, K_IN, K_H);
    gemm_bf16<<<dim3(K_TOK / 128, K_IN / 32), dim3(64), 0, stream>>>(xn, up_t, nullptr, ubuf, K_TOK, K_IN, K_H);
    silu_mul_kernel<<<dim3((K_TOK * K_IN + 255) / 256), dim3(256), 0, stream>>>(
        gbuf, ubuf, interb, K_TOK * K_IN);
    gemm_bf16<<<dim3(K_TOK / 128, K_H / 32), dim3(64), 0, stream>>>(
        interb, down_t, x, x, K_TOK, K_H, K_IN);  // residual add into x
  }

  // ---- head ----
  head_kernel<<<dim3(K_B), dim3(256), 0, stream>>>(
      x, final_nw, act_w, act_b, act_high, act_low, out);
}
